// FAME_7361573945548
// MI455X (gfx1250) — matrix operand order repaired and checked by reading, not compiled, not run on hardware
//
#include <hip/hip_runtime.h>
#include <hip/hip_bf16.h>
#include <math.h>

// ---------------- problem constants ----------------
#define BB    8
#define CCH   3
#define TT    16
#define HH    224
#define WWD   224
#define HWX   50176          // 224*224
#define NSLOT 9              // 1 im_diff + 8 frame-pair diffs
#define NIMG  72             // NSLOT * BB
#define KSZ   23
#define PADR  11
#define SIGG  (23.0f/3.0f)
#define EPSF  1e-8f
#define KFG   25088          // int(0.5*HW)
#define KBG   5017           // int(0.1*HW)
#define NUMFG 25088
#define NBIN  1024
#define MDICT 1000
#define VIDN  19267584       // 8*3*16*224*224
#define PERCLIP (CCH*TT*HWX) // elems per batch item in videos

typedef __attribute__((ext_vector_type(16))) _Float16 v16h;
typedef __attribute__((ext_vector_type(8)))  _Float16 v8h;
typedef __attribute__((ext_vector_type(8)))  float    v8f;
typedef __attribute__((ext_vector_type(4)))  float    v4f;

// jax.random.permutation(jax.random.key(1), 8) -- fixed, hardcoded
__constant__ int c_perm[8] = {6, 0, 3, 5, 7, 2, 4, 1};

// ---------------- optional Tensor Data Mover staging ----------------
#if __has_builtin(__builtin_amdgcn_tensor_load_to_lds) && __has_builtin(__builtin_amdgcn_s_wait_tensorcnt)
#define HAVE_TDM 1
typedef unsigned int u32x4 __attribute__((ext_vector_type(4)));
typedef int          i32x4 __attribute__((ext_vector_type(4)));
typedef int          i32x8 __attribute__((ext_vector_type(8)));

// DMA a 32x32 f16 tile (row stride 224 elements) from global into LDS (packed rows).
__device__ __forceinline__ void tdm_load_tile_f16(const _Float16* gsrc, unsigned ldsOff) {
  unsigned long long ga = (unsigned long long)(size_t)gsrc;
  u32x4 g0;
  g0[0] = 1u;                                              // count=1, user descriptor
  g0[1] = ldsOff;                                          // lds_addr (bytes)
  g0[2] = (unsigned)(ga & 0xffffffffu);                    // global_addr[31:0]
  g0[3] = (unsigned)((ga >> 32) & 0x01ffffffu) | (2u << 30); // addr[56:32] | type=2
  i32x8 g1;
  g1[0] = (int)(1u << 16);          // workgroup_mask=0 | data_size=1 (2 bytes)
  g1[1] = (int)(224u << 16);        // tensor_dim0 low16 in [31:16]
  g1[2] = (int)(224u << 16);        // tensor_dim0 hi=0 | tensor_dim1 low16
  g1[3] = (int)(32u << 16);         // tensor_dim1 hi=0 | tile_dim0=32
  g1[4] = 32;                       // tile_dim1=32, tile_dim2=0
  g1[5] = 224;                      // tensor_dim0_stride low32
  g1[6] = (int)(50176u << 16);      // stride0 hi=0 | tensor_dim1_stride low16
  g1[7] = 0;                        // tensor_dim1_stride hi
  i32x4 gz4 = {0, 0, 0, 0};
#if defined(__clang_major__) && __clang_major__ >= 23
  i32x8 gz8 = {0, 0, 0, 0, 0, 0, 0, 0};
  __builtin_amdgcn_tensor_load_to_lds(g0, g1, gz4, gz4, gz8, 0);
#else
  __builtin_amdgcn_tensor_load_to_lds(g0, g1, gz4, gz4, 0);
#endif
}
#else
#define HAVE_TDM 0
#endif

// ---------------- build reflect-padded Gaussian blur matrix G (224x224, f16) ----------------
__global__ void build_G(_Float16* G) {
  int i = blockIdx.x, j = threadIdx.x;        // grid 224, block 224
  float w[KSZ]; float s = 0.f;
  #pragma unroll
  for (int d = 0; d < KSZ; ++d) {
    float x = (float)d - (float)PADR;
    w[d] = __expf(-(x * x) / (2.f * SIGG * SIGG));
    s += w[d];
  }
  float acc = 0.f;
  #pragma unroll
  for (int d = 0; d < KSZ; ++d) {
    int r = i - PADR + d;
    if (r < 0) r = -r;
    if (r > HH - 1) r = 2 * (HH - 1) - r;     // 'reflect' pad
    if (r == j) acc += w[d] / s;
  }
  G[i * HH + j] = (_Float16)acc;
}

// ---------------- one pass over videos: diffs (f16, straight into GEMM input) + cmap ----------------
__global__ void compute_sources(const float* __restrict__ vid,
                                _Float16* __restrict__ diffh,  // [NSLOT][BB][HWX]
                                int* __restrict__ cmap) {      // [BB][HWX]
  int idx = blockIdx.x * blockDim.x + threadIdx.x;
  if (idx >= BB * HWX) return;
  int b = idx / HWX, pix = idx % HWX;
  const float stdv[3]  = {0.229f, 0.224f, 0.225f};
  const float meanv[3] = {0.485f, 0.456f, 0.406f};
  float pair[8]; float mrgb[3]; float imdiff = 0.f;
  #pragma unroll
  for (int p = 0; p < 8; ++p) pair[p] = 0.f;
  for (int c = 0; c < CCH; ++c) {
    size_t base = ((size_t)(b * CCH + c) * TT) * HWX + pix;
    float sumv = 0.f, dsum = 0.f, prev = 0.f;
    for (int t = 0; t < TT; ++t) {
      float v = vid[base + (size_t)t * HWX];
      sumv += v;
      if (t > 0) dsum += fabsf(v - prev);
      if (t & 1) pair[t >> 1] += stdv[c] * fabsf(v - prev);
      prev = v;
    }
    imdiff += stdv[c] * dsum * (1.f / 15.f);
    mrgb[c] = stdv[c] * (sumv * (1.f / 16.f)) + meanv[c];
  }
  diffh[(size_t)b * HWX + pix] = (_Float16)imdiff;                     // slot 0
  #pragma unroll
  for (int p = 0; p < 8; ++p)
    diffh[((size_t)(1 + p) * BB + b) * HWX + pix] = (_Float16)pair[p]; // slots 1..8
  // --- rgb_to_hsv + color-cube index (matches reference math, incl. double 2*pi) ---
  float r = mrgb[0], g = mrgb[1], bl = mrgb[2];
  float maxc = fmaxf(r, fmaxf(g, bl));
  float minc = fminf(r, fminf(g, bl));
  float v = maxc, deltac = maxc - minc;
  float sat = deltac / (v + EPSF);
  float dc = (deltac == 0.f) ? 1.f : deltac;
  float rc = (maxc - r) / dc, gc = (maxc - g) / dc, bc = (maxc - bl) / dc;
  float h = (maxc == r) ? (bc - gc) : ((maxc == g) ? (2.f + rc - bc) : (4.f + gc - rc));
  h = h * (1.f / 6.f);
  h = h - floorf(h);
  h = h * 6.28318530717958647692f;            // img_h in [0, 2pi]
  float ang = h * 6.28318530717958647692f;    // reference multiplies by 2*pi again
  float hx = (sat * __cosf(ang) + 1.f) * 0.5f;
  float hy = (sat * __sinf(ang) + 1.f) * 0.5f;
  float hb = rintf(hx * 9.f + 1.f);
  float sb = rintf(hy * 9.f + 1.f);
  float vb = rintf(v * 9.f + 1.f);
  int cm = (int)(hb + (sb - 1.f) * 10.f + (vb - 1.f) * 100.f);
  cmap[idx] = min(max(cm, 0), MDICT - 1);
}

// ---------------- WMMA half-blur: OUT = (Z * G^T)^T ----------------
// Applying this twice yields G*X*G^T (full separable blur).
// Both operands are K-contiguous row-major (A = Z rows, B-col n = G row n), so
// staging is two straight 2D tile copies (TDM or coalesced b128), no transposes.
// The output transpose is free: each lane stores 8 consecutive elements.
// Block = 128 threads (4 waves) -> 32x32 output region (2x2 tiles of 16x16).
template <bool OUTHALF>
__global__ void wmma_halfblur(const _Float16* __restrict__ Zbase,
                              const _Float16* __restrict__ G,
                              void* __restrict__ Cbase,
                              int zStride, int cStride) {
  __shared__ _Float16 As[32][32];   // As[m][k] = Z[brow0+m][k0+k]
  __shared__ _Float16 Bs[32][32];   // Bs[n][k] = G[bcol0+n][k0+k]
  int blk = blockIdx.x;
  int img = blk / 49, t = blk % 49;
  int brow0 = (t / 7) * 32, bcol0 = (t % 7) * 32;
  const _Float16* Z = Zbase + (size_t)img * zStride;
  int tid  = threadIdx.x;
  int wave = tid >> 5, lane = tid & 31;
  int wr = wave >> 1, wc = wave & 1;            // 2x2 tile grid inside block
  int ln15 = lane & 15, half = lane >> 4;
  int srow = tid >> 2, sseg = tid & 3;          // staging: 32 rows x 4 x 8-half segments
#if HAVE_TDM
  unsigned asOff = (unsigned)(size_t)&As[0][0]; // flat LDS addr[31:0] == LDS offset
  unsigned bsOff = (unsigned)(size_t)&Bs[0][0];
#endif
  v8f acc = {};
  for (int k0 = 0; k0 < 224; k0 += 32) {
#if HAVE_TDM
    if (wave == 0) {                            // TDM: EXEC-independent, once per block
      tdm_load_tile_f16(&Z[(size_t)brow0 * 224 + k0], asOff);
      tdm_load_tile_f16(&G[(size_t)bcol0 * 224 + k0], bsOff);
      __builtin_amdgcn_s_wait_tensorcnt(0);
    }
#else
    *(v8h*)&As[srow][sseg * 8] =
        *(const v8h*)&Z[(size_t)(brow0 + srow) * 224 + k0 + sseg * 8];
    *(v8h*)&Bs[srow][sseg * 8] =
        *(const v8h*)&G[(size_t)(bcol0 + srow) * 224 + k0 + sseg * 8];
#endif
    __syncthreads();
    // ---- fragments from LDS (all contiguous 16B reads) ----
    int arow = wr * 16 + ln15;
    v16h a = *(const v16h*)&As[arow][half * 16];       // K = half*16 .. +15
    int bcol = wc * 16 + ln15;
    v16h bf = *(const v16h*)&Bs[bcol][half * 16];      // K = half*16 .. +15
    acc = __builtin_amdgcn_wmma_f32_16x16x32_f16(
        /*neg_a=*/false, a, /*neg_b=*/false, bf,
        /*c_mod=*/(short)0, acc, /*reuse_a=*/false, /*reuse_b=*/false);
    __syncthreads();
  }
  // ---- transposed store: lane writes 8 consecutive elements ----
  int row0 = brow0 + wr * 16, col0 = bcol0 + wc * 16;
  int ocol = col0 + ln15;
  size_t obase = (size_t)img * cStride + (size_t)ocol * 224 + row0 + (half << 3);
  if (OUTHALF) {
    v8h hv;
    #pragma unroll
    for (int r = 0; r < 8; ++r) hv[r] = (_Float16)acc[r];
    *(v8h*)((_Float16*)Cbase + obase) = hv;
  } else {
    v4f lo4 = {acc[0], acc[1], acc[2], acc[3]};
    v4f hi4 = {acc[4], acc[5], acc[6], acc[7]};
    *(v4f*)((float*)Cbase + obase)     = lo4;
    *(v4f*)((float*)Cbase + obase + 4) = hi4;
  }
}

// ---------------- per-image min/max ----------------
__global__ void minmax_img(const float* __restrict__ x, float* __restrict__ mm) {
  __shared__ float smin[256], smax[256];
  int img = blockIdx.x, tid = threadIdx.x;
  const float* p = x + (size_t)img * HWX;
  float lo = 1e30f, hi = -1e30f;
  for (int i = tid; i < HWX; i += 256) {
    float v = p[i]; lo = fminf(lo, v); hi = fmaxf(hi, v);
  }
  smin[tid] = lo; smax[tid] = hi; __syncthreads();
  for (int s = 128; s > 0; s >>= 1) {
    if (tid < s) { smin[tid] = fminf(smin[tid], smin[tid + s]);
                   smax[tid] = fmaxf(smax[tid], smax[tid + s]); }
    __syncthreads();
  }
  if (tid == 0) { mm[img * 2] = smin[0]; mm[img * 2 + 1] = smax[0]; }
}

// ---------------- normalize in place + histogram ----------------
__global__ void norm_hist(float* __restrict__ x, const float* __restrict__ mm,
                          unsigned* __restrict__ hist) {
  int i = blockIdx.x * blockDim.x + threadIdx.x;
  if (i >= NIMG * HWX) return;
  int img = i / HWX;
  float lo = mm[img * 2], hi = mm[img * 2 + 1];
  float v = (x[i] - lo) / ((hi - lo) + EPSF);
  x[i] = v;
  int bin = min(NBIN - 1, max(0, (int)(v * (float)NBIN)));
  atomicAdd(&hist[img * NBIN + bin], 1u);
}

// ---------------- histogram -> value thresholds (replaces top_k) ----------------
__global__ void find_thresholds(const unsigned* __restrict__ hist, float* __restrict__ thr,
                                int kfg, int kbg) {
  int img = blockIdx.x * blockDim.x + threadIdx.x;
  if (img >= NIMG) return;
  const unsigned* h = hist + img * NBIN;
  int cum = 0; float tfg = 0.f;
  for (int b = NBIN - 1; b >= 0; --b) {
    cum += (int)h[b];
    if (cum >= kfg) { tfg = (float)b / (float)NBIN; break; }
  }
  int cum2 = 0; float tbg = 1.f;
  for (int b = 0; b < NBIN; ++b) {
    cum2 += (int)h[b];
    if (cum2 >= kbg) { tbg = (float)(b + 1) / (float)NBIN; break; }
  }
  thr[img * 2] = tfg; thr[img * 2 + 1] = tbg;
}

// ---------------- fg/bg color dictionaries ----------------
__global__ void build_dicts(const float* __restrict__ nrm, const int* __restrict__ cmap,
                            const float* __restrict__ thr,
                            float* __restrict__ dfg, float* __restrict__ dbg) {
  int i = blockIdx.x * blockDim.x + threadIdx.x;
  if (i >= NIMG * HWX) return;
  int img = i / HWX, pix = i % HWX, b = img % BB;
  float v = nrm[i];
  int c = cmap[b * HWX + pix];
  if (v >= thr[img * 2])     atomicAdd(&dfg[img * MDICT + c], 1.f);
  if (v <= thr[img * 2 + 1]) atomicAdd(&dbg[img * MDICT + c], 1.f);
}

__global__ void dict_sums(const float* __restrict__ dfg, const float* __restrict__ dbg,
                          float* __restrict__ dsum) {
  __shared__ float sf[256], sb[256];
  int img = blockIdx.x, tid = threadIdx.x;
  float a = 0.f, b = 0.f;
  for (int i = tid; i < MDICT; i += 256) {
    a += dfg[img * MDICT + i]; b += dbg[img * MDICT + i];
  }
  sf[tid] = a; sb[tid] = b; __syncthreads();
  for (int s = 128; s > 0; s >>= 1) {
    if (tid < s) { sf[tid] += sf[tid + s]; sb[tid] += sb[tid + s]; }
    __syncthreads();
  }
  if (tid == 0) { dsum[img * 2] = sf[0]; dsum[img * 2 + 1] = sb[0]; }
}

// refine written as f16 directly (input of the stage-B blur GEMM)
__global__ void refine_k(const float* __restrict__ dfg, const float* __restrict__ dbg,
                         const float* __restrict__ dsum, const int* __restrict__ cmap,
                         _Float16* __restrict__ out) {
  int i = blockIdx.x * blockDim.x + threadIdx.x;
  if (i >= NIMG * HWX) return;
  int img = i / HWX, pix = i % HWX, b = img % BB;
  int c = cmap[b * HWX + pix];
  float pf = dfg[img * MDICT + c] / (dsum[img * 2] + EPSF);
  float pb = (dbg[img * MDICT + c] + 1.f) / (dsum[img * 2 + 1] + (float)MDICT + EPSF);
  out[i] = (_Float16)(pf / (pb + pf));
}

__global__ void binarize(const float* __restrict__ nrm, const float* __restrict__ thr,
                         float* __restrict__ masks) {
  int i = blockIdx.x * blockDim.x + threadIdx.x;
  if (i >= NIMG * HWX) return;
  int img = i / HWX;
  masks[i] = (nrm[i] >= thr[img * 2]) ? 1.f : 0.f;
}

// ---------------- 16x16 avg pool for mask_out + mpf_out ----------------
__global__ void pool_outputs(const float* __restrict__ masks,
                             float* __restrict__ out_mask, float* __restrict__ out_mpf) {
  int i = blockIdx.x * blockDim.x + threadIdx.x;   // 14112 total
  if (i >= BB * 196 + BB * 8 * 196) return;
  int img, oy, ox, oi; float* dst;
  if (i < BB * 196) {
    int b = i / 196, r = i % 196; oy = r / 14; ox = r % 14;
    img = b; dst = out_mask; oi = i;               // slot 0
  } else {
    int j = i - BB * 196;
    int b = j / (8 * 196), rem = j % (8 * 196);
    int p = rem / 196, r = rem % 196; oy = r / 14; ox = r % 14;
    img = (1 + p) * BB + b; dst = out_mpf; oi = j;
  }
  const float* mp = masks + (size_t)img * HWX;
  float s = 0.f;
  for (int y = 0; y < 16; ++y)
    for (int x = 0; x < 16; ++x)
      s += mp[(oy * 16 + y) * WWD + ox * 16 + x];
  dst[oi] = s * (1.f / 256.f);
}

// ---------------- video fuse ----------------
__global__ void fuse_k(const float* __restrict__ vid, const float* __restrict__ masks,
                       float* __restrict__ out) {
  int idx = blockIdx.x * blockDim.x + threadIdx.x;
  if (idx >= VIDN) return;
  int pix = idx % HWX;
  int b   = idx / PERCLIP;
  int rem = idx % PERCLIP;
  const float* shp = vid + (size_t)c_perm[b] * PERCLIP + rem;
  __builtin_prefetch(shp, 0, 1);                   // global_prefetch_b8 on the shuffled stream
  float m = masks[(size_t)b * HWX + pix];          // slot-0 binary mask
  float own = vid[idx];
  float sh  = *shp;
  out[idx] = sh * (1.f - m) + own * m;
}

__global__ void label_k(const int* __restrict__ lab, float* __restrict__ out) {
  int i = threadIdx.x;
  if (i < BB) out[i] = (float)lab[i];
}

// ---------------- host orchestration ----------------
static inline size_t alignUp(size_t x) { return (x + 255) & ~(size_t)255; }

static void run_blur(const _Float16* xh, _Float16* vh, float* blur,
                     const _Float16* G, float* mm, unsigned* hist, hipStream_t stream) {
  const int nelem = NIMG * HWX;
  // R(Z) = (Z*G^T)^T applied twice == G*X*G^T
  wmma_halfblur<true ><<<NIMG * 49, 128, 0, stream>>>(xh, G, vh,   HWX, HWX);
  wmma_halfblur<false><<<NIMG * 49, 128, 0, stream>>>(vh, G, blur, HWX, HWX);
  minmax_img<<<NIMG, 256, 0, stream>>>(blur, mm);
  hipMemsetAsync(hist, 0, (size_t)NIMG * NBIN * sizeof(unsigned), stream);
  norm_hist<<<(nelem + 255) / 256, 256, 0, stream>>>(blur, mm, hist);
}

extern "C" void kernel_launch(void* const* d_in, const int* in_sizes, int n_in,
                              void* d_out, int out_size, void* d_ws, size_t ws_size,
                              hipStream_t stream) {
  const float* vid = (const float*)d_in[0];
  const int*   lab = (const int*)d_in[1];
  float* out = (float*)d_out;

  // workspace carve-up
  char* w = (char*)d_ws; size_t off = 0;
  _Float16* G    = (_Float16*)(w + off); off = alignUp(off + (size_t)HWX * 2);
  _Float16* xh   = (_Float16*)(w + off); off = alignUp(off + (size_t)NIMG * HWX * 2); // GEMM input (diff/refine)
  _Float16* vh   = (_Float16*)(w + off); off = alignUp(off + (size_t)NIMG * HWX * 2); // intermediate
  float*    blur = (float*)(w + off);    off = alignUp(off + (size_t)NIMG * HWX * 4); // normalized in place
  float*    mask = (float*)(w + off);    off = alignUp(off + (size_t)NIMG * HWX * 4);
  int*      cmap = (int*)(w + off);      off = alignUp(off + (size_t)BB * HWX * 4);
  float*    mm   = (float*)(w + off);    off = alignUp(off + (size_t)NIMG * 2 * 4);
  unsigned* hist = (unsigned*)(w + off); off = alignUp(off + (size_t)NIMG * NBIN * 4);
  float*    thr  = (float*)(w + off);    off = alignUp(off + (size_t)NIMG * 2 * 4);
  float*    dfg  = (float*)(w + off);    off = alignUp(off + (size_t)NIMG * MDICT * 4);
  float*    dbg  = (float*)(w + off);    off = alignUp(off + (size_t)NIMG * MDICT * 4);
  float*    dsum = (float*)(w + off);    off = alignUp(off + (size_t)NIMG * 2 * 4);
  (void)ws_size; (void)in_sizes; (void)n_in; (void)out_size;

  const int nelem = NIMG * HWX;
  const int nblk  = (nelem + 255) / 256;

  build_G<<<224, 224, 0, stream>>>(G);
  compute_sources<<<(BB * HWX + 255) / 256, 256, 0, stream>>>(vid, xh, cmap);

  // ---- segmentation stage A: blur(diff) -> norm -> fg/bg thresholds -> dicts -> refine ----
  run_blur(xh, vh, blur, G, mm, hist, stream);
  find_thresholds<<<1, 128, 0, stream>>>(hist, thr, KFG, KBG);
  hipMemsetAsync(dfg, 0, (size_t)NIMG * MDICT * 4, stream);
  hipMemsetAsync(dbg, 0, (size_t)NIMG * MDICT * 4, stream);
  build_dicts<<<nblk, 256, 0, stream>>>(blur, cmap, thr, dfg, dbg);
  dict_sums<<<NIMG, 256, 0, stream>>>(dfg, dbg, dsum);
  refine_k<<<nblk, 256, 0, stream>>>(dfg, dbg, dsum, cmap, xh);   // xh reused as refine (f16)

  // ---- stage B: blur(refine) -> norm -> top-50% binary mask ----
  run_blur(xh, vh, blur, G, mm, hist, stream);
  find_thresholds<<<1, 128, 0, stream>>>(hist, thr, NUMFG, 1);
  binarize<<<nblk, 256, 0, stream>>>(blur, thr, mask);

  // ---- outputs: video_fuse | label | mask_out | mpf_out ----
  fuse_k<<<VIDN / 256, 256, 0, stream>>>(vid, mask, out);
  label_k<<<1, 32, 0, stream>>>(lab, out + VIDN);
  pool_outputs<<<(14112 + 255) / 256, 256, 0, stream>>>(
      mask, out + VIDN + BB, out + VIDN + BB + BB * 196);
}
